// HeadPac_43164421325454
// MI455X (gfx1250) — matrix operand order repaired and checked by reading, not compiled, not run on hardware
//
#include <hip/hip_runtime.h>
#include <hip/hip_bf16.h>

typedef __attribute__((ext_vector_type(16))) _Float16 v16h;
typedef __attribute__((ext_vector_type(8)))  float    v8f;
typedef __attribute__((ext_vector_type(4)))  unsigned int u32x4;
typedef __attribute__((ext_vector_type(8)))  int          i32x8;
typedef __attribute__((ext_vector_type(4)))  int          i32x4;

// ---------------------------------------------------------------------------
// conv1_1: Ci=1 (K=9) -> plain VALU kernel, WMMA would waste 3/4 of K.
// ---------------------------------------------------------------------------
__global__ void __launch_bounds__(256) conv3x3_c1(
    const float* __restrict__ x, const float* __restrict__ w,
    const float* __restrict__ bias, float* __restrict__ out,
    int B, int Co, int H, int W) {
  int idx = blockIdx.x * blockDim.x + threadIdx.x;
  int total = B * Co * H * W;
  if (idx >= total) return;
  int wx = idx % W;
  int hy = (idx / W) % H;
  int co = (idx / (W * H)) % Co;
  int b  = idx / (W * H * Co);
  const float* xb = x + (size_t)b * H * W;
  const float* wk = w + co * 9;
  float acc = bias[co];
#pragma unroll
  for (int di = 0; di < 3; ++di)
#pragma unroll
    for (int dj = 0; dj < 3; ++dj) {
      int ih = hy + di - 1, iw = wx + dj - 1;
      float v = (ih >= 0 && ih < H && iw >= 0 && iw < W) ? xb[ih * W + iw] : 0.f;
      acc = fmaf(wk[di * 3 + dj], v, acc);
    }
  out[idx] = fmaxf(acc, 0.f);
}

// ---------------------------------------------------------------------------
// PAC gaussian kernel: kout[b,tap,h,w] = exp(-0.5 * sum_c (s*g_nb - s*g_ctr)^2)
// ---------------------------------------------------------------------------
__global__ void __launch_bounds__(256) pac_gauss(
    const float* __restrict__ g, float* __restrict__ kout,
    float scale, int B, int C, int H, int W) {
  int idx = blockIdx.x * blockDim.x + threadIdx.x;
  int total = B * H * W;
  if (idx >= total) return;
  int wx = idx % W;
  int hy = (idx / W) % H;
  int b  = idx / (W * H);
  float s[9];
#pragma unroll
  for (int t = 0; t < 9; ++t) s[t] = 0.f;
  for (int c = 0; c < C; ++c) {
    const float* gc = g + (((size_t)b * C + c) * H) * W;
    float ctr = gc[hy * W + wx] * scale;
#pragma unroll
    for (int di = 0; di < 3; ++di)
#pragma unroll
      for (int dj = 0; dj < 3; ++dj) {
        int ih = hy + di - 1, iw = wx + dj - 1;
        float nb = (ih >= 0 && ih < H && iw >= 0 && iw < W) ? gc[ih * W + iw] * scale : 0.f;
        float d = nb - ctr;
        s[di * 3 + dj] = fmaf(d, d, s[di * 3 + dj]);
      }
  }
#pragma unroll
  for (int t = 0; t < 9; ++t)
    kout[(((size_t)b * 9 + t) * H + hy) * W + wx] = __expf(-0.5f * s[t]);
}

// ---------------------------------------------------------------------------
// Repack OIHW f32 weights into the per-lane WMMA A-fragment image (f16):
//   wp[(((tile*nCk + ck)*9 + tap)*32 + lane)*16 + e]
// ---------------------------------------------------------------------------
__global__ void __launch_bounds__(256) pack_weights(
    const float* __restrict__ w, _Float16* __restrict__ wp, int Co, int Ci) {
  int idx = blockIdx.x * blockDim.x + threadIdx.x;
  int total = Co * Ci * 9;
  if (idx >= total) return;
  int tap = idx % 9;
  int ci  = (idx / 9) % Ci;
  int co  = idx / (9 * Ci);
  int tile = co >> 4, m = co & 15;
  int ck = ci >> 5, cl = ci & 31;
  int g = cl >> 4;
  int e = cl & 15;
  int lane = m + 16 * g;
  int nCk = Ci >> 5;
  wp[((((size_t)tile * nCk + ck) * 9 + tap) * 32 + lane) * 16 + e] =
      (_Float16)w[idx];
}

// ---------------------------------------------------------------------------
// TDM: 2D tile (rows x elems0 of 8-byte elements) -> LDS.  D# per ISA 8.3-8.6.
// group0: dw0 count=1 | dw1 lds_addr | dw2..3 global_addr(57b) + type=2
// group1: dw0 data_size=8B | td0/td1 | tile_dim0/1 | dim0 stride
// ---------------------------------------------------------------------------
__device__ __forceinline__ unsigned lds_off(const void* p) {
  // generic LDS address: addr[31:0] is the wave-relative LDS byte offset
  return (unsigned)(unsigned long long)(uintptr_t)p;
}
__device__ __forceinline__ void tdm_load_2d(unsigned ldsAddr, const void* gsrc,
                                            unsigned elems0, unsigned rows,
                                            unsigned long long rowStride) {
  unsigned long long ga = (unsigned long long)(uintptr_t)gsrc;
  u32x4 g0;
  g0.x = 1u;                                           // count=1 (valid D#)
  g0.y = ldsAddr;                                      // lds_addr (bytes)
  g0.z = (unsigned)ga;                                 // global_addr[31:0]
  g0.w = ((unsigned)(ga >> 32) & 0x01ffffffu) | (2u << 30);  // [56:32] | type=2
  i32x8 g1;
  g1[0] = (int)(3u << 16);                             // data_size = 8 bytes
  g1[1] = (int)((elems0 & 0xffffu) << 16);             // tensor_dim0[15:0]
  g1[2] = (int)((elems0 >> 16) & 0xffffu)              // tensor_dim0[31:16]
        | (int)((rows & 0xffffu) << 16);               // tensor_dim1[15:0]
  g1[3] = (int)((elems0 & 0xffffu) << 16);             // td1 hi=0 | tile_dim0
  g1[4] = (int)(rows & 0xffffu);                       // tile_dim1 | tile_dim2=0
  g1[5] = (int)(rowStride & 0xffffffffull);            // dim0_stride[31:0]
  g1[6] = (int)((rowStride >> 32) & 0xffffull);        // dim0_stride[47:32]
  g1[7] = 0;                                           // dim1_stride unused (2D)
  i32x4 z4 = {0, 0, 0, 0};
  i32x8 z8 = {0, 0, 0, 0, 0, 0, 0, 0};
  __builtin_amdgcn_tensor_load_to_lds(g0, g1, z4, z4, z8, 0);
}

// ---------------------------------------------------------------------------
// Implicit-GEMM (PAC-)conv via v_wmma_f32_16x16x32_f16.
// Block: 4 waves = 64 Co x 16 pixels.  Weights stream via TDM into LDS
// (issued for ck+1 after compute, waited on before next chunk's staging);
// activations staged as f16 channel-innermost slab, reused by all 9 taps.
// ---------------------------------------------------------------------------
#define NW 4
#define SLAB_HALF (3 * 36 * 32)                    // 6912 B
#define WCHUNK_HALF (NW * 9 * 32 * 16)             // 36864 B
template <int STRIDE>
__global__ void __launch_bounds__(32 * NW) pacconv_wmma(
    const float* __restrict__ x, const float* __restrict__ kern,
    const _Float16* __restrict__ wp, const float* __restrict__ bias,
    float* __restrict__ out, int B, int Ci, int Co, int H, int W) {
  const int oH = H / STRIDE, oW = W / STRIDE;
  const int tid  = threadIdx.x;
  const int wave = tid >> 5;
  const int lane = tid & 31;
  const int n = lane & 15;
  const int g = lane >> 4;
  const int owBase = blockIdx.x * 16;
  const int oh     = blockIdx.y;
  const int coTiles = Co >> 6;
  const int zt     = blockIdx.z % coTiles;
  const int b      = blockIdx.z / coTiles;
  const int coBase = zt * 64 + wave * 16;
  constexpr int WIN = 16 * STRIDE + 2;             // 18 or 34
  extern __shared__ char smem[];
  _Float16* slab = (_Float16*)smem;                           // f16 slab
  _Float16* wlds = (_Float16*)(smem + SLAB_HALF * 2);         // weight panel

  v8f c = {};
  const int nCk = Ci >> 5;

  // hoist PAC modulation scalars (loop-invariant over K-chunks)
  _Float16 kvs[9];
#pragma unroll
  for (int t = 0; t < 9; ++t) kvs[t] = (_Float16)1.f;
  if (kern) {
#pragma unroll
    for (int t = 0; t < 9; ++t)
      kvs[t] = (_Float16)kern[(((size_t)b * 9 + t) * H + oh * STRIDE) * W +
                              (owBase + n) * STRIDE];
  }

  // TDM descriptor constants: 4 rows (one per wave tile) of 9216 B
  const unsigned wldsAddr = lds_off(wlds);
  const unsigned long long rowStride = (unsigned long long)nCk * 1152ull;
  auto wsrc = [&](int ck) {
    return (const void*)(wp + ((size_t)(zt * 4) * nCk + ck) * 4608);
  };

  if (wave == 0) tdm_load_2d(wldsAddr, wsrc(0), 1152u, NW, rowStride);

  for (int ck = 0; ck < nCk; ++ck) {
    if (wave == 0) __builtin_amdgcn_s_wait_tensorcnt(0);  // weights landed
    __syncthreads();                                      // + slab WAR
    // stage [3][WIN][32] f16 slab (zero-padded), prefetch next chunk
    const int tot = 32 * 3 * WIN;
    for (int i = tid; i < tot; i += 32 * NW) {
      int col = i % WIN;
      int t2  = i / WIN;
      int r   = t2 % 3;
      int cl  = t2 / 3;
      int ih = oh * STRIDE - 1 + r;
      int iw = owBase * STRIDE - 1 + col;
      float v = 0.f;
      if (ih >= 0 && ih < H && iw >= 0 && iw < W) {
        const float* p = &x[(((size_t)b * Ci + ck * 32 + cl) * H + ih) * W + iw];
        v = *p;
        if (ck + 1 < nCk) __builtin_prefetch(p + (size_t)32 * H * W, 0, 1);
      }
      slab[(r * 36 + col) * 32 + cl] = (_Float16)v;
    }
    __syncthreads();

    const _Float16* wl = wlds + wave * 4608 + lane * 16;
#pragma unroll
    for (int di = 0; di < 3; ++di) {
#pragma unroll
      for (int dj = 0; dj < 3; ++dj) {
        const int tap = di * 3 + dj;
        const int col = n * STRIDE + dj;
        v16h bh = *(const v16h*)&slab[(di * 36 + col) * 32 + 16 * g];
        v16h bv = bh * kvs[tap];                       // v_pk_mul_f16 x8
        v16h av = *(const v16h*)(wl + tap * 512);      // 2 x ds_load_b128
        c = __builtin_amdgcn_wmma_f32_16x16x32_f16(
            false, av, false, bv, (short)0, c, false, false);
      }
    }
    if (ck + 1 < nCk) {
      __syncthreads();                                 // all reads of wlds done
      if (wave == 0)
        tdm_load_2d(wldsAddr, wsrc(ck + 1), 1152u, NW, rowStride);
    }
  }

  // C/D layout: element k at lane l -> M = k + 8g, N = n
#pragma unroll
  for (int k = 0; k < 8; ++k) {
    int co = coBase + k + 8 * g;
    float v = c[k] + bias[co];
    v = fmaxf(v, 0.f);
    out[(((size_t)b * Co + co) * oH + oh) * oW + owBase + n] = v;
  }
}

// ---------------------------------------------------------------------------
extern "C" void kernel_launch(void* const* d_in, const int* in_sizes, int n_in,
                              void* d_out, int out_size, void* d_ws, size_t ws_size,
                              hipStream_t stream) {
  const float* x = (const float*)d_in[0];
  const float* W[14];
  const float* Bs[14];
  for (int i = 0; i < 14; ++i) {
    W[i]  = (const float*)d_in[1 + 2 * i];
    Bs[i] = (const float*)d_in[2 + 2 * i];
  }
  static const int CoA[14] = {64, 64, 64, 128, 128, 128, 256, 256, 256,
                              512, 512, 512, 1024, 1024};
  static const int CiA[14] = {1, 64, 64, 64, 128, 128, 128, 256, 256,
                              256, 512, 512, 512, 1024};

  float* out = (float*)d_out;
  const size_t off_h1 = 0;
  const size_t off_h2 = off_h1 + (size_t)2 * 64 * 256 * 256;
  const size_t off_h3 = off_h2 + (size_t)2 * 128 * 128 * 128;
  const size_t off_h4 = off_h3 + (size_t)2 * 256 * 64 * 64;
  const size_t off_h5 = off_h4 + (size_t)2 * 512 * 32 * 32;
  const size_t off_k2 = off_h5 + (size_t)2 * 1024 * 16 * 16;
  const size_t off_k3 = off_k2 + (size_t)2 * 9 * 128 * 128;
  const size_t off_k4 = off_k3 + (size_t)2 * 9 * 64 * 64;
  const size_t off_k5 = off_k4 + (size_t)2 * 9 * 32 * 32;

  // ws layout: [G arena][T arena][packed f16 weights]
  float* wsG = (float*)d_ws;
  float* wsT = wsG + (size_t)2 * 64 * 128 * 128;
  _Float16* packBase = (_Float16*)(wsT + (size_t)2 * 64 * 256 * 256);

  _Float16* WP[14];
  {
    size_t off = 0;
    for (int li = 1; li < 14; ++li) {
      WP[li] = packBase + off;
      int total = CoA[li] * CiA[li] * 9;
      pack_weights<<<(total + 255) / 256, 256, 0, stream>>>(W[li], WP[li],
                                                            CoA[li], CiA[li]);
      off += (size_t)total;
    }
  }

  const float KC = 1e-4f;
  const size_t smemBytes = SLAB_HALF * 2 + WCHUNK_HALF * 2;  // 43776 B

  auto conv = [&](const float* in, const float* kern, int li, float* o,
                  int B, int H, int Wd, int s) {
    dim3 grid((Wd / s) / 16, H / s, B * (CoA[li] / 64));
    if (s == 1)
      pacconv_wmma<1><<<grid, 32 * NW, smemBytes, stream>>>(
          in, kern, WP[li], Bs[li], o, B, CiA[li], CoA[li], H, Wd);
    else
      pacconv_wmma<2><<<grid, 32 * NW, smemBytes, stream>>>(
          in, kern, WP[li], Bs[li], o, B, CiA[li], CoA[li], H, Wd);
  };
  auto gauss = [&](const float* gp, float* o, float sc, int B, int C, int H, int Wd) {
    int total = B * H * Wd;
    pac_gauss<<<(total + 255) / 256, 256, 0, stream>>>(gp, o, sc, B, C, H, Wd);
  };

  // stage 1
  {
    int total = 2 * 64 * 256 * 256;
    conv3x3_c1<<<(total + 255) / 256, 256, 0, stream>>>(x, W[0], Bs[0], wsT,
                                                        2, 64, 256, 256);
  }
  conv(wsT, nullptr, 1, out + off_h1, 2, 256, 256, 1);              // 1_2 -> h1
  conv(out + off_h1, nullptr, 2, wsG, 2, 256, 256, 2);              // 1_3 -> g2

  // stage 2
  gauss(wsG, out + off_k2, KC, 2, 64, 128, 128);                    // k2
  conv(wsG, out + off_k2, 3, wsT, 2, 128, 128, 1);                  // 2_1
  conv(wsT, out + off_k2, 4, out + off_h2, 2, 128, 128, 1);         // 2_2 -> h2
  conv(out + off_h2, out + off_k2, 5, wsG, 2, 128, 128, 2);         // 2_3 -> g3

  // stage 3
  gauss(wsG, out + off_k3, KC, 2, 128, 64, 64);                     // k3
  conv(wsG, out + off_k3, 6, wsT, 2, 64, 64, 1);                    // 3_1
  conv(wsT, out + off_k3, 7, out + off_h3, 2, 64, 64, 1);           // 3_2 -> h3
  conv(out + off_h3, out + off_k3, 8, wsG, 2, 64, 64, 2);           // 3_3 -> g4

  // stage 4 (guidance scale = 2.0 in the source)
  gauss(wsG, out + off_k4, 2.0f, 2, 256, 32, 32);                   // k4
  conv(wsG, out + off_k4, 9, wsT, 2, 32, 32, 1);                    // 4_1
  conv(wsT, out + off_k4, 10, out + off_h4, 2, 32, 32, 1);          // 4_2 -> h4
  conv(out + off_h4, out + off_k4, 11, wsG, 2, 32, 32, 2);          // 4_3 -> g5

  // stage 5
  gauss(wsG, out + off_k5, KC, 2, 512, 16, 16);                     // k5
  conv(wsG, out + off_k5, 12, wsT, 2, 16, 16, 1);                   // 5_1
  conv(wsT, out + off_k5, 13, out + off_h5, 2, 16, 16, 1);          // 5_2 -> h5
}
